// MambaForSummarization_7421703488074
// MI455X (gfx1250) — compile-verified, matrix operand order checked by reading, not run on hardware
//
#include <hip/hip_runtime.h>
#include <hip/hip_bf16.h>

// ---------------------------------------------------------------------------
// MambaForSummarization forward on MI455X (gfx1250, wave32, WMMA bf16 path)
// ---------------------------------------------------------------------------

typedef __attribute__((ext_vector_type(16))) __bf16 v16bf;
typedef __attribute__((ext_vector_type(8)))  float  v8f;

#define D_    512
#define DI_   1024
#define DS_   16
#define NL_   6
#define SRC_  1024
#define TGT_  256
#define VOC_  30522
#define HEADS_ 8
#define HD_   64

enum { ACT_NONE = 0, ACT_SILU = 1, ACT_SOFTPLUS = 2 };

__device__ __forceinline__ float silu_f(float x)     { return x / (1.0f + __expf(-x)); }
__device__ __forceinline__ float softplus_f(float x) { return (x > 20.0f) ? x : log1pf(__expf(x)); }

// ---------------------------------------------------------------------------
// Embedding gather: out[t, :] = embed[ids[t], :]
// ---------------------------------------------------------------------------
__global__ __launch_bounds__(256)
void embed_kernel(const int* __restrict__ ids, const float* __restrict__ emb,
                  float* __restrict__ out)
{
    int i = blockIdx.x * 256 + threadIdx.x;      // tokens * 512 elements
    int t = i >> 9;
    int c = i & 511;
    out[i] = emb[(size_t)ids[t] * D_ + c];
}

// ---------------------------------------------------------------------------
// LayerNorm over last dim (512), one block (256 threads) per row
// ---------------------------------------------------------------------------
__global__ __launch_bounds__(256)
void ln_kernel(const float* __restrict__ x, const float* __restrict__ s,
               const float* __restrict__ b, float* __restrict__ out)
{
    __shared__ float rsum[256];
    __shared__ float rsq[256];
    const int row = blockIdx.x;
    const int tid = threadIdx.x;
    const float* xr = x + (size_t)row * D_;
    float v0 = xr[tid];
    float v1 = xr[tid + 256];
    rsum[tid] = v0 + v1;
    rsq[tid]  = v0 * v0 + v1 * v1;
    __syncthreads();
    for (int off = 128; off > 0; off >>= 1) {
        if (tid < off) { rsum[tid] += rsum[tid + off]; rsq[tid] += rsq[tid + off]; }
        __syncthreads();
    }
    const float mean = rsum[0] * (1.0f / D_);
    const float var  = rsq[0] * (1.0f / D_) - mean * mean;
    const float inv  = rsqrtf(var + 1e-6f);
    out[(size_t)row * D_ + tid]       = (v0 - mean) * inv * s[tid]       + b[tid];
    out[(size_t)row * D_ + tid + 256] = (v1 - mean) * inv * s[tid + 256] + b[tid + 256];
}

// ---------------------------------------------------------------------------
// Generic GEMM:  C[M,N] = act(A[M,K] @ B[K,N] + bias)  (optional += residual)
// f32 in memory, bf16 WMMA (v_wmma_f32_16x16x32_bf16), f32 accumulate.
// 128 threads = 4 waves; each wave owns one 16x16 tile; blockIdx.y = M tile.
// M, K multiples of 16/32 at all call sites; N may be ragged (LM head).
// Ragged-N handling is branch-free in the K loop: OOB lanes load a clamped
// in-bounds column (their accumulators are simply never stored), so EXEC
// stays full and no per-iteration exec save/restore is generated.
// ---------------------------------------------------------------------------
__global__ __launch_bounds__(128)
void gemm_bf16_kernel(const float* __restrict__ A, const float* __restrict__ B,
                      const float* __restrict__ bias, float* __restrict__ C,
                      int M, int N, int K, int act, int addTo)
{
    const int lane = threadIdx.x & 31;
    const int wave = threadIdx.x >> 5;
    const int tn   = blockIdx.x * 4 + wave;
    if (tn * 16 >= N) return;                         // wave-uniform exit
    const int tm   = blockIdx.y;
    const int half = lane >> 4;                       // 0: lanes 0-15, 1: 16-31
    const int r16  = lane & 15;
    const int grow = tm * 16 + r16;                   // A row for this lane
    const int gcol = tn * 16 + r16;                   // B/D column for this lane
    const bool colOK = (gcol < N);
    const int  gc    = colOK ? gcol : (N - 1);        // clamped, always in-bounds

    v8f acc;
    {
        float bv = bias ? bias[gc] : 0.0f;
        #pragma unroll
        for (int r = 0; r < 8; ++r) acc[r] = bv;
    }

    const size_t stepN = (size_t)N;
    const float* Ap = A + (size_t)grow * K + 8 * half;   // this lane's A stream
    const float* Bp = B + (size_t)(8 * half) * stepN + gc; // this lane's B stream

    for (int k0 = 0; k0 < K; k0 += 32) {
        v16bf a, b;
        // A fragment per ISA 16-bit 16x32 layout: this lane holds K = kb..kb+7
        // and kb+16..kb+23 of row `grow` (kb = k0 + 8*half).
        const float4 a0 = *(const float4*)(Ap);
        const float4 a1 = *(const float4*)(Ap + 4);
        const float4 a2 = *(const float4*)(Ap + 16);
        const float4 a3 = *(const float4*)(Ap + 20);
        a[0]=(__bf16)a0.x;  a[1]=(__bf16)a0.y;  a[2]=(__bf16)a0.z;  a[3]=(__bf16)a0.w;
        a[4]=(__bf16)a1.x;  a[5]=(__bf16)a1.y;  a[6]=(__bf16)a1.z;  a[7]=(__bf16)a1.w;
        a[8]=(__bf16)a2.x;  a[9]=(__bf16)a2.y;  a[10]=(__bf16)a2.z; a[11]=(__bf16)a2.w;
        a[12]=(__bf16)a3.x; a[13]=(__bf16)a3.y; a[14]=(__bf16)a3.z; a[15]=(__bf16)a3.w;
        #pragma unroll
        for (int j = 0; j < 8; ++j) {
            b[j]     = (__bf16)Bp[(size_t)j * stepN];
            b[j + 8] = (__bf16)Bp[(size_t)(j + 16) * stepN];
        }
        if (k0 + 32 < K) __builtin_prefetch(Bp + 32 * stepN, 0, 1);
        Ap += 32;
        Bp += 32 * stepN;
        acc = __builtin_amdgcn_wmma_f32_16x16x32_bf16(false, a, false, b,
                                                      (short)0, acc, false, false);
    }

    if (colOK) {
        #pragma unroll
        for (int r = 0; r < 8; ++r) {
            float v = acc[r];
            if (act == ACT_SILU)          v = silu_f(v);
            else if (act == ACT_SOFTPLUS) v = softplus_f(v);
            size_t idx = (size_t)(tm * 16 + 8 * half + r) * stepN + gcol;
            C[idx] = addTo ? (C[idx] + v) : v;
        }
    }
}

// ---------------------------------------------------------------------------
// Depth-dense conv1d (SAME, width 4) as a GEMM with K = 4*1024.
// A[l, kk] = XZ[l-1+(kk>>10), kk&1023] (row stride 2*DI, zero-padded),
// B = conv_w viewed as (4096, 1024) row-major.  Output = silu(conv + bias).
// Boundary rows are handled branch-free: load a clamped in-bounds row, then
// value-select zero (v_cndmask), keeping EXEC full for the WMMA.
// ---------------------------------------------------------------------------
__global__ __launch_bounds__(128)
void conv_gemm_kernel(const float* __restrict__ XZ, const float* __restrict__ W,
                      const float* __restrict__ bias, float* __restrict__ out, int L)
{
    const int lane = threadIdx.x & 31;
    const int wave = threadIdx.x >> 5;
    const int tn   = blockIdx.x * 4 + wave;     // N tile (N = 1024, always full)
    const int tm   = blockIdx.y;
    const int half = lane >> 4;
    const int r16  = lane & 15;
    const int grow = tm * 16 + r16;
    const int gcol = tn * 16 + r16;

    v8f acc;
    {
        float bv = bias[gcol];
        #pragma unroll
        for (int r = 0; r < 8; ++r) acc[r] = bv;
    }

    const float* Bp = W + (size_t)(8 * half) * DI_ + gcol;  // B stream

    for (int k0 = 0; k0 < 4 * DI_; k0 += 32) {
        const int kb = k0 + 8 * half;
        // 8-wide K chunks never straddle a 1024 boundary (kb multiple of 8)
        const int w0 = kb >> 10,        i0 = kb & 1023;
        const int w1 = (kb + 16) >> 10, i1 = (kb + 16) & 1023;
        int sr0 = grow - 1 + w0;
        int sr1 = grow - 1 + w1;
        const bool ok0 = (sr0 >= 0) && (sr0 < L);
        const bool ok1 = (sr1 >= 0) && (sr1 < L);
        sr0 = min(max(sr0, 0), L - 1);              // clamped: loads in-bounds
        sr1 = min(max(sr1, 0), L - 1);
        const float* p0 = XZ + (size_t)sr0 * (2 * DI_) + i0;
        const float* p1 = XZ + (size_t)sr1 * (2 * DI_) + i1;
        const float4 a0 = *(const float4*)(p0);
        const float4 a1 = *(const float4*)(p0 + 4);
        const float4 a2 = *(const float4*)(p1);
        const float4 a3 = *(const float4*)(p1 + 4);
        const float m0 = ok0 ? 1.0f : 0.0f;
        const float m1 = ok1 ? 1.0f : 0.0f;
        v16bf a, b;
        a[0]=(__bf16)(a0.x*m0);  a[1]=(__bf16)(a0.y*m0);
        a[2]=(__bf16)(a0.z*m0);  a[3]=(__bf16)(a0.w*m0);
        a[4]=(__bf16)(a1.x*m0);  a[5]=(__bf16)(a1.y*m0);
        a[6]=(__bf16)(a1.z*m0);  a[7]=(__bf16)(a1.w*m0);
        a[8]=(__bf16)(a2.x*m1);  a[9]=(__bf16)(a2.y*m1);
        a[10]=(__bf16)(a2.z*m1); a[11]=(__bf16)(a2.w*m1);
        a[12]=(__bf16)(a3.x*m1); a[13]=(__bf16)(a3.y*m1);
        a[14]=(__bf16)(a3.z*m1); a[15]=(__bf16)(a3.w*m1);
        #pragma unroll
        for (int j = 0; j < 8; ++j) {
            b[j]     = (__bf16)Bp[(size_t)j * DI_];
            b[j + 8] = (__bf16)Bp[(size_t)(j + 16) * DI_];
        }
        if (k0 + 32 < 4 * DI_) __builtin_prefetch(Bp + (size_t)32 * DI_, 0, 1);
        Bp += (size_t)32 * DI_;
        acc = __builtin_amdgcn_wmma_f32_16x16x32_bf16(false, a, false, b,
                                                      (short)0, acc, false, false);
    }

    #pragma unroll
    for (int r = 0; r < 8; ++r) {
        float v = silu_f(acc[r]);
        out[(size_t)(tm * 16 + 8 * half + r) * DI_ + gcol] = v;
    }
}

// ---------------------------------------------------------------------------
// Selective-scan:  h[l,n,s] = sum_{j<=l} delta[j,n]*B[j,s]*xc[j,n]*0.95^(j+1)
//                  y[l,n]   = sum_s h[l,n,s] * C[l,s];   out = y * silu(z)
// One thread per channel n (1024 total), 16-state recurrence in registers.
// ---------------------------------------------------------------------------
__global__ __launch_bounds__(256)
void scan_kernel(const float* __restrict__ xz, const float* __restrict__ xc,
                 const float* __restrict__ delta, const float* __restrict__ bm,
                 const float* __restrict__ cm, float* __restrict__ yg, int L)
{
    const int n = blockIdx.x * 256 + threadIdx.x;   // 0..DI_-1
    float S[DS_];
    #pragma unroll
    for (int s = 0; s < DS_; ++s) S[s] = 0.0f;
    float dec = 1.0f;
    for (int l = 0; l < L; ++l) {
        dec *= 0.95f;
        const float coeff = delta[(size_t)l * DI_ + n] * xc[(size_t)l * DI_ + n] * dec;
        const float* bl = bm + (size_t)l * DS_;
        const float* cl = cm + (size_t)l * DS_;
        float yv = 0.0f;
        #pragma unroll
        for (int s = 0; s < DS_; ++s) {
            S[s] += coeff * bl[s];
            yv   += S[s] * cl[s];
        }
        const float z = xz[(size_t)l * (2 * DI_) + DI_ + n];
        yg[(size_t)l * DI_ + n] = yv * silu_f(z);
    }
}

// ---------------------------------------------------------------------------
// Cross-attention: one block (64 threads) per (target token t, head h).
// scores over S=1024 encoder tokens -> softmax -> ctx[64].
// ---------------------------------------------------------------------------
__global__ __launch_bounds__(64)
void attn_kernel(const float* __restrict__ Q, const float* __restrict__ Kb,
                 const float* __restrict__ Vb, const int* __restrict__ mask,
                 float* __restrict__ CTX)
{
    __shared__ float sc[SRC_];
    __shared__ float qs[HD_];
    __shared__ float red[64];
    const int t = blockIdx.x >> 3;
    const int h = blockIdx.x & 7;
    const int d = threadIdx.x;
    qs[d] = Q[(size_t)t * D_ + h * HD_ + d];
    __syncthreads();
    float lmax = -1e30f;
    for (int s = d; s < SRC_; s += 64) {
        const float* kp = Kb + (size_t)s * D_ + h * HD_;
        float dot = 0.0f;
        #pragma unroll 8
        for (int dd = 0; dd < HD_; ++dd) dot += qs[dd] * kp[dd];
        dot *= 0.125f;                       // 1/sqrt(64)
        if (mask[s] == 0) dot = -10000.0f;
        sc[s] = dot;
        lmax = fmaxf(lmax, dot);
    }
    red[d] = lmax;
    __syncthreads();
    for (int off = 32; off > 0; off >>= 1) {
        if (d < off) red[d] = fmaxf(red[d], red[d + off]);
        __syncthreads();
    }
    const float m = red[0];
    __syncthreads();
    float lsum = 0.0f;
    for (int s = d; s < SRC_; s += 64) {
        float e = __expf(sc[s] - m);
        sc[s] = e;
        lsum += e;
    }
    red[d] = lsum;
    __syncthreads();
    for (int off = 32; off > 0; off >>= 1) {
        if (d < off) red[d] += red[d + off];
        __syncthreads();
    }
    const float inv = 1.0f / red[0];
    __syncthreads();
    float acc = 0.0f;
    for (int s = 0; s < SRC_; ++s)
        acc += sc[s] * Vb[(size_t)s * D_ + h * HD_ + d];
    CTX[(size_t)t * D_ + h * HD_ + d] = acc * inv;
}

// ---------------------------------------------------------------------------
// Host-side orchestration
// ---------------------------------------------------------------------------
extern "C" void kernel_launch(void* const* d_in, const int* in_sizes, int n_in,
                              void* d_out, int out_size, void* d_ws, size_t ws_size,
                              hipStream_t stream)
{
    (void)in_sizes; (void)n_in; (void)out_size; (void)ws_size;

    const int*   input_ids = (const int*)d_in[0];
    const int*   attn_mask = (const int*)d_in[1];
    const int*   dec_ids   = (const int*)d_in[2];
    const float* embed     = (const float*)d_in[3];
    const float* enc_ln_s  = (const float*)d_in[4];
    const float* enc_ln_b  = (const float*)d_in[5];
    const float* enc_in_w  = (const float*)d_in[6];
    const float* enc_cv_w  = (const float*)d_in[7];
    const float* enc_cv_b  = (const float*)d_in[8];
    const float* enc_dt_w  = (const float*)d_in[9];
    const float* enc_dt_b  = (const float*)d_in[10];
    const float* enc_B_w   = (const float*)d_in[11];
    const float* enc_B_b   = (const float*)d_in[12];
    const float* enc_C_w   = (const float*)d_in[13];
    const float* enc_C_b   = (const float*)d_in[14];
    /* d_in[15] = enc_A : dead code in reference */
    const float* enc_out_w = (const float*)d_in[16];
    const float* enc_fln_s = (const float*)d_in[17];
    const float* enc_fln_b = (const float*)d_in[18];
    const float* dec_ln1_s = (const float*)d_in[19];
    const float* dec_ln1_b = (const float*)d_in[20];
    const float* dec_in_w  = (const float*)d_in[21];
    const float* dec_cv_w  = (const float*)d_in[22];
    const float* dec_cv_b  = (const float*)d_in[23];
    const float* dec_dt_w  = (const float*)d_in[24];
    const float* dec_dt_b  = (const float*)d_in[25];
    const float* dec_B_w   = (const float*)d_in[26];
    const float* dec_B_b   = (const float*)d_in[27];
    const float* dec_C_w   = (const float*)d_in[28];
    const float* dec_C_b   = (const float*)d_in[29];
    /* d_in[30] = dec_A : dead code */
    const float* dec_out_w = (const float*)d_in[31];
    const float* dec_ln2_s = (const float*)d_in[32];
    const float* dec_ln2_b = (const float*)d_in[33];
    const float* q_w = (const float*)d_in[34];
    const float* q_b = (const float*)d_in[35];
    const float* k_w = (const float*)d_in[36];
    const float* k_b = (const float*)d_in[37];
    const float* v_w = (const float*)d_in[38];
    const float* v_b = (const float*)d_in[39];
    const float* o_w = (const float*)d_in[40];
    const float* o_b = (const float*)d_in[41];
    const float* dec_fln_s = (const float*)d_in[42];
    const float* dec_fln_b = (const float*)d_in[43];
    const float* lm_w = (const float*)d_in[44];
    const float* lm_b = (const float*)d_in[45];

    // workspace layout (floats)
    float* ws   = (float*)d_ws;
    float* X    = ws;                     // SRC*D      = 524288
    float* XN   = X    + 524288;          // SRC*D
    float* XZ   = XN   + 524288;          // SRC*2DI    = 2097152
    float* XC   = XZ   + 2097152;         // SRC*DI     = 1048576
    float* DEL  = XC   + 1048576;         // SRC*DI
    float* BMb  = DEL  + 1048576;         // SRC*DS     = 16384
    float* CMb  = BMb  + 16384;           // SRC*DS
    float* YG   = CMb  + 16384;           // SRC*DI
    float* ENC  = YG   + 1048576;         // SRC*D
    float* Y    = ENC  + 524288;          // TGT*D      = 131072
    float* Qb   = Y    + 131072;          // TGT*D
    float* Kbf  = Qb   + 131072;          // SRC*D
    float* Vbf  = Kbf  + 524288;          // SRC*D
    float* CTX  = Vbf  + 524288;          // TGT*D

    auto gemm = [&](const float* A, const float* B, const float* bias, float* C,
                    int M, int N, int K, int act, int addTo) {
        dim3 grid(((N + 15) / 16 + 3) / 4, M / 16);
        gemm_bf16_kernel<<<grid, dim3(128), 0, stream>>>(A, B, bias, C, M, N, K, act, addTo);
    };

    // ---------------- encoder ----------------
    embed_kernel<<<SRC_ * D_ / 256, 256, 0, stream>>>(input_ids, embed, X);

    for (int i = 0; i < NL_; ++i) {
        ln_kernel<<<SRC_, 256, 0, stream>>>(X, enc_ln_s + i * D_, enc_ln_b + i * D_, XN);
        gemm(XN, enc_in_w + (size_t)i * D_ * 2 * DI_, nullptr, XZ,
             SRC_, 2 * DI_, D_, ACT_NONE, 0);
        conv_gemm_kernel<<<dim3(DI_ / 64, SRC_ / 16), dim3(128), 0, stream>>>(
            XZ, enc_cv_w + (size_t)i * 4 * DI_ * DI_, enc_cv_b + i * DI_, XC, SRC_);
        gemm(XC, enc_dt_w + (size_t)i * DI_ * DI_, enc_dt_b + i * DI_, DEL,
             SRC_, DI_, DI_, ACT_SOFTPLUS, 0);
        gemm(XC, enc_B_w + (size_t)i * DI_ * DS_, enc_B_b + i * DS_, BMb,
             SRC_, DS_, DI_, ACT_NONE, 0);
        gemm(XC, enc_C_w + (size_t)i * DI_ * DS_, enc_C_b + i * DS_, CMb,
             SRC_, DS_, DI_, ACT_NONE, 0);
        scan_kernel<<<DI_ / 256, 256, 0, stream>>>(XZ, XC, DEL, BMb, CMb, YG, SRC_);
        gemm(YG, enc_out_w + (size_t)i * DI_ * D_, nullptr, X,
             SRC_, D_, DI_, ACT_NONE, 1);                    // residual add
    }
    ln_kernel<<<SRC_, 256, 0, stream>>>(X, enc_fln_s, enc_fln_b, ENC);

    // ---------------- decoder ----------------
    embed_kernel<<<TGT_ * D_ / 256, 256, 0, stream>>>(dec_ids, embed, Y);

    for (int i = 0; i < NL_; ++i) {
        // mamba block
        ln_kernel<<<TGT_, 256, 0, stream>>>(Y, dec_ln1_s + i * D_, dec_ln1_b + i * D_, XN);
        gemm(XN, dec_in_w + (size_t)i * D_ * 2 * DI_, nullptr, XZ,
             TGT_, 2 * DI_, D_, ACT_NONE, 0);
        conv_gemm_kernel<<<dim3(DI_ / 64, TGT_ / 16), dim3(128), 0, stream>>>(
            XZ, dec_cv_w + (size_t)i * 4 * DI_ * DI_, dec_cv_b + i * DI_, XC, TGT_);
        gemm(XC, dec_dt_w + (size_t)i * DI_ * DI_, dec_dt_b + i * DI_, DEL,
             TGT_, DI_, DI_, ACT_SOFTPLUS, 0);
        gemm(XC, dec_B_w + (size_t)i * DI_ * DS_, dec_B_b + i * DS_, BMb,
             TGT_, DS_, DI_, ACT_NONE, 0);
        gemm(XC, dec_C_w + (size_t)i * DI_ * DS_, dec_C_b + i * DS_, CMb,
             TGT_, DS_, DI_, ACT_NONE, 0);
        scan_kernel<<<DI_ / 256, 256, 0, stream>>>(XZ, XC, DEL, BMb, CMb, YG, TGT_);
        gemm(YG, dec_out_w + (size_t)i * DI_ * D_, nullptr, Y,
             TGT_, D_, DI_, ACT_NONE, 1);                    // residual add

        // cross-attention block
        ln_kernel<<<TGT_, 256, 0, stream>>>(Y, dec_ln2_s + i * D_, dec_ln2_b + i * D_, XN);
        gemm(XN,  q_w + (size_t)i * D_ * D_, q_b + i * D_, Qb,  TGT_, D_, D_, ACT_NONE, 0);
        gemm(ENC, k_w + (size_t)i * D_ * D_, k_b + i * D_, Kbf, SRC_, D_, D_, ACT_NONE, 0);
        gemm(ENC, v_w + (size_t)i * D_ * D_, v_b + i * D_, Vbf, SRC_, D_, D_, ACT_NONE, 0);
        attn_kernel<<<TGT_ * HEADS_, 64, 0, stream>>>(Qb, Kbf, Vbf, attn_mask, CTX);
        gemm(CTX, o_w + (size_t)i * D_ * D_, o_b + i * D_, Y,
             TGT_, D_, D_, ACT_NONE, 1);                     // residual add
    }

    // ---------------- LM head ----------------
    ln_kernel<<<TGT_, 256, 0, stream>>>(Y, dec_fln_s, dec_fln_b, XN);
    gemm(XN, lm_w, lm_b, (float*)d_out, TGT_, VOC_, D_, ACT_NONE, 0);
}
